// MultiLatentAttention_57526791962859
// MI455X (gfx1250) — matrix operand order repaired and checked by reading, not compiled, not run on hardware
//
#include <hip/hip_runtime.h>

// ---------------------------------------------------------------------------
// Multi-Latent Attention for MI455X (gfx1250), bf16 WMMA pipeline.
//   B=2 T=2048 C=1024 H=16 HD=64 L=8 LD=128, S=T+L=2056 (keys padded to 2080)
// Latent *queries* are dropped by the reference output slice -> never computed.
// ---------------------------------------------------------------------------

typedef unsigned short u16;
typedef unsigned int   u32;
typedef __attribute__((ext_vector_type(16))) __bf16 v16bf;
typedef __attribute__((ext_vector_type(8)))  float  v8f;

#define Bc  2
#define Tc  2048
#define Cc  1024
#define Hc  16
#define HDc 64
#define Lc  8
#define LDc 128
#define Sc  2056      // real keys + latent keys
#define SPc 2080      // padded key count (multiple of 32)

union FragB { v16bf v; u32 u[8]; };

__device__ __forceinline__ u16 f2bf(float f) {           // fp32 -> bf16 RNE
  u32 x = __float_as_uint(f);
  return (u16)((x + 0x7FFFu + ((x >> 16) & 1u)) >> 16);
}

// 16-bit A-matrix (16x32) per-lane K index for VGPR r (contiguous: k = 16*half + 2r)
__device__ __forceinline__ int kmapA(int r, int half) {
  return (half << 4) + (r << 1);
}

__device__ __forceinline__ v8f wmma_bf16(const FragB& a, const FragB& b, v8f c) {
  return __builtin_amdgcn_wmma_f32_16x16x32_bf16(false, a.v, false, b.v,
                                                 (short)0, c, false, false);
}

// ---------------------------------------------------------------- converters
__global__ __launch_bounds__(256) void cvt_bf16(const float* __restrict__ s,
                                                u16* __restrict__ d, int n) {
  int i = blockIdx.x * 256 + threadIdx.x;
  if (i < n) d[i] = f2bf(s[i]);
}

// W (K x N) fp32 row-major  ->  Wt (N x K) bf16 row-major
__global__ __launch_bounds__(256) void cvtT_bf16(const float* __restrict__ w,
                                                 u16* __restrict__ wt,
                                                 int Kd, int N) {
  int i = blockIdx.x * 256 + threadIdx.x;
  if (i < Kd * N) {
    int n = i / Kd, k = i % Kd;
    wt[i] = f2bf(w[(size_t)k * N + n]);
  }
}

__global__ __launch_bounds__(256) void zero_u16(u16* __restrict__ d, int n) {
  int i = blockIdx.x * 256 + threadIdx.x;
  if (i < n) d[i] = 0;
}

// latent keys: lk = lat_k(L x LD) @ Wlk(LD x C)  -> K rows T..T+L-1, both b
__global__ __launch_bounds__(256) void latk_kernel(const float* __restrict__ lk,
                                                   const float* __restrict__ Wlk,
                                                   u16* __restrict__ Kb) {
  int i = blockIdx.x * 256 + threadIdx.x;           // L*C = 8192
  if (i >= Lc * Cc) return;
  int l = i >> 10, c = i & (Cc - 1);
  float s = 0.f;
  for (int d = 0; d < LDc; ++d) s += lk[l * LDc + d] * Wlk[(size_t)d * Cc + c];
  u16 bv = f2bf(s);
  int h = c >> 6, hd = c & 63;
  for (int b = 0; b < Bc; ++b)
    Kb[(((size_t)(b * Hc + h) * SPc) + Tc + l) * HDc + hd] = bv;
}

// ------------------------------------------------------------------ GEMM
// A (M x K) bf16 row-major, Bt (N x K) bf16 (pre-transposed weights),
// one 16x64 tile per wave, K stepped by 32 with v_wmma_f32_16x16x32_bf16.
// ep: 0 = fp32 store (final proj), 1 = RoPE->Q pack, 2 = RoPE->K pack,
//     3 = transpose store -> Vt (B,H,HD,SPc)
__global__ __launch_bounds__(256) void gemm_bf16_ep(const u16* __restrict__ A,
                                                    const u16* __restrict__ Bt,
                                                    float* __restrict__ Cf,
                                                    u16* __restrict__ Dst,
                                                    int M, int N, int Kd, int ep) {
  const int wid = threadIdx.x >> 5;
  const int lane = threadIdx.x & 31;
  const int half = lane >> 4, lh = lane & 15;
  const int gw = blockIdx.x * 8 + wid;
  const int nT = N >> 6;
  const int mt = gw / nT, nt = gw % nT;

  const u16* Ap = A + (size_t)(mt * 16 + lh) * Kd;
  v8f acc[4] = {};

  for (int kk = 0; kk < Kd; kk += 32) {
    FragB a;
#pragma unroll
    for (int r = 0; r < 8; ++r)
      a.u[r] = *(const u32*)(Ap + kk + kmapA(r, half));
#pragma unroll
    for (int t = 0; t < 4; ++t) {
      FragB b;
      const u16* Bp = Bt + (size_t)(nt * 64 + t * 16 + lh) * Kd + kk + (half << 4);
#pragma unroll
      for (int r = 0; r < 8; ++r) b.u[r] = *(const u32*)(Bp + (r << 1));
      acc[t] = wmma_bf16(a, b, acc[t]);
    }
  }

  if (ep == 0) {
#pragma unroll
    for (int t = 0; t < 4; ++t)
#pragma unroll
      for (int r = 0; r < 8; ++r) {
        int row = mt * 16 + r + (half << 3);
        int col = nt * 64 + t * 16 + lh;
        Cf[(size_t)row * N + col] = acc[t][r];
      }
  } else if (ep == 3) {                       // V -> (B,H,HD,SPc) bf16
#pragma unroll
    for (int t = 0; t < 4; ++t)
#pragma unroll
      for (int r = 0; r < 8; ++r) {
        int row = mt * 16 + r + (half << 3);
        int col = nt * 64 + t * 16 + lh;
        int b = row >> 11, tt = row & (Tc - 1);
        int h = col >> 6, hd = col & 63;
        Dst[((size_t)(b * Hc + h) * HDc + hd) * SPc + tt] = f2bf(acc[t][r]);
      }
  } else {                                    // RoPE then pack Q (ep1) / K (ep2)
#pragma unroll
    for (int t = 0; t < 4; ++t)
#pragma unroll
      for (int r = 0; r < 8; ++r) {
        float v = acc[t][r];
        float p = __shfl_xor(v, 1);           // RoPE partner (adjacent column)
        int row = mt * 16 + r + (half << 3);
        int col = nt * 64 + t * 16 + lh;
        int b = row >> 11, tt = row & (Tc - 1);
        int h = col >> 6, hd = col & 63;
        // freq = 10000^(-(2i)/64), 2i = hd & ~1 ; log2(10000) = 13.2877124
        float freq = exp2f(-(float)(hd & ~1) * (13.28771238f / 64.0f));
        float ang = (float)tt * freq;
        float sn, cs;
        __sincosf(ang, &sn, &cs);
        float rv = (hd & 1) ? (p * sn + v * cs) : (v * cs - p * sn);
        size_t off = (ep == 1)
                         ? (((size_t)(b * Hc + h) * Tc + tt) * HDc + hd)
                         : (((size_t)(b * Hc + h) * SPc + tt) * HDc + hd);
        Dst[off] = f2bf(rv);
      }
  }
}

// -------------------------------------------------------------- attention
// One wave per 16-query tile of one (b,h). Flash-style online softmax.
// Qb (B,H,T,64) bf16, Kb (B,H,SPc,64) bf16, Vt (B,H,64,SPc) bf16,
// Ob (B,T,C) bf16.
__global__ __launch_bounds__(128) void attn_kernel(const u16* __restrict__ Qb,
                                                   const u16* __restrict__ Kb,
                                                   const u16* __restrict__ Vt,
                                                   u16* __restrict__ Ob) {
  __shared__ u16 Ps[4][16][32];               // per-wave P staging (C->A layout)

  const int wid = threadIdx.x >> 5;
  const int lane = threadIdx.x & 31;
  const int half = lane >> 4, lh = lane & 15;
  const int gw = blockIdx.x * 4 + wid;        // 4096 waves total
  const int tq = gw & 127;                    // T/16 = 128 query tiles
  const int bh = gw >> 7;                     // b*H + h
  const int tq16 = tq * 16;

  const u16* Qp = Qb + ((size_t)bh * Tc + tq16) * HDc;
  const u16* Kp = Kb + (size_t)bh * SPc * HDc;
  const u16* Vp = Vt + (size_t)bh * HDc * SPc;

  FragB aQ0, aQ1;                             // Q tile, hd 0..31 / 32..63
#pragma unroll
  for (int r = 0; r < 8; ++r) {
    int ko = kmapA(r, half);
    aQ0.u[r] = *(const u32*)(Qp + (size_t)lh * HDc + ko);
    aQ1.u[r] = *(const u32*)(Qp + (size_t)lh * HDc + 32 + ko);
  }

  v8f acc[4] = {};
  float mrow[8], lrow[8];
#pragma unroll
  for (int r = 0; r < 8; ++r) { mrow[r] = -1e30f; lrow[r] = 0.f; }

  const float scale = 0.125f;                 // 1/sqrt(64)
  const int jend = (tq16 + 16 + 31) & ~31;    // causal coverage, rounded to 32

  for (int phase = 0; phase < 2; ++phase) {
    const int jlo = phase ? Tc : 0;
    const int jhi = phase ? SPc : jend;
    for (int j0 = jlo; j0 < jhi; j0 += 32) {
      // ---- scores: two 16-key chunks, K-dim 64 via two bf16 WMMAs each
      v8f c0 = {}, c1 = {};
      {
        FragB bK;
#pragma unroll
        for (int r = 0; r < 8; ++r)
          bK.u[r] = *(const u32*)(Kp + (size_t)(j0 + lh) * HDc + (half << 4) + (r << 1));
        c0 = wmma_bf16(aQ0, bK, c0);
#pragma unroll
        for (int r = 0; r < 8; ++r)
          bK.u[r] = *(const u32*)(Kp + (size_t)(j0 + lh) * HDc + 32 + (half << 4) + (r << 1));
        c0 = wmma_bf16(aQ1, bK, c0);
#pragma unroll
        for (int r = 0; r < 8; ++r)
          bK.u[r] = *(const u32*)(Kp + (size_t)(j0 + 16 + lh) * HDc + (half << 4) + (r << 1));
        c1 = wmma_bf16(aQ0, bK, c1);
#pragma unroll
        for (int r = 0; r < 8; ++r)
          bK.u[r] = *(const u32*)(Kp + (size_t)(j0 + 16 + lh) * HDc + 32 + (half << 4) + (r << 1));
        c1 = wmma_bf16(aQ1, bK, c1);
      }

      // ---- mask + online softmax (row = r + 8*half in tile)
      const int k0 = j0 + lh, k1 = j0 + 16 + lh;
#pragma unroll
      for (int r = 0; r < 8; ++r) {
        int rowg = tq16 + r + (half << 3);
        bool ok0 = (k0 <= rowg) || (k0 >= Tc && k0 < Sc);
        bool ok1 = (k1 <= rowg) || (k1 >= Tc && k1 < Sc);
        float s0 = ok0 ? c0[r] * scale : -1e30f;
        float s1 = ok1 ? c1[r] * scale : -1e30f;
        float mx = fmaxf(s0, s1);
        mx = fmaxf(mx, __shfl_xor(mx, 1));
        mx = fmaxf(mx, __shfl_xor(mx, 2));
        mx = fmaxf(mx, __shfl_xor(mx, 4));
        mx = fmaxf(mx, __shfl_xor(mx, 8));
        float mnew = fmaxf(mrow[r], mx);
        float corr = __expf(mrow[r] - mnew);
        float p0 = __expf(s0 - mnew);
        float p1 = __expf(s1 - mnew);
        float rs = p0 + p1;
        rs += __shfl_xor(rs, 1);
        rs += __shfl_xor(rs, 2);
        rs += __shfl_xor(rs, 4);
        rs += __shfl_xor(rs, 8);
        lrow[r] = lrow[r] * corr + rs;
        mrow[r] = mnew;
        acc[0][r] *= corr; acc[1][r] *= corr;
        acc[2][r] *= corr; acc[3][r] *= corr;
        int m = r + (half << 3);
        Ps[wid][m][lh] = f2bf(p0);            // C-layout -> row-major LDS
        Ps[wid][m][16 + lh] = f2bf(p1);
      }

      asm volatile("s_wait_dscnt 0x0" ::: "memory");   // LDS store->load fence

      FragB aP;                               // reload P in A-fragment layout
#pragma unroll
      for (int r = 0; r < 8; ++r)
        aP.u[r] = *(const u32*)&Ps[wid][lh][kmapA(r, half)];

      // ---- O += P(16x32) x V(32x64), V transposed so pairs are key-contig.
#pragma unroll
      for (int t = 0; t < 4; ++t) {
        FragB bV;
#pragma unroll
        for (int r = 0; r < 8; ++r)
          bV.u[r] = *(const u32*)(Vp + (size_t)(t * 16 + lh) * SPc + j0 + (half << 4) + (r << 1));
        acc[t] = wmma_bf16(aP, bV, acc[t]);
      }
    }
  }

  // ---- normalize and store into (B,T,C) bf16 layout for the output GEMM
  const int b = bh >> 4, h = bh & 15;
#pragma unroll
  for (int t = 0; t < 4; ++t)
#pragma unroll
    for (int r = 0; r < 8; ++r) {
      int row = tq16 + r + (half << 3);
      int col = h * HDc + t * 16 + lh;
      Ob[((size_t)b * Tc + row) * Cc + col] = f2bf(acc[t][r] / lrow[r]);
    }
}

// ------------------------------------------------------------------ launch
extern "C" void kernel_launch(void* const* d_in, const int* in_sizes, int n_in,
                              void* d_out, int out_size, void* d_ws, size_t ws_size,
                              hipStream_t stream) {
  const float* x     = (const float*)d_in[0];
  const float* Wq    = (const float*)d_in[1];
  const float* Wk    = (const float*)d_in[2];
  const float* Wv    = (const float*)d_in[3];
  // d_in[4] lat_q and d_in[6] Wlq are dead: latent query rows are dropped.
  const float* lat_k = (const float*)d_in[5];
  const float* Wlk   = (const float*)d_in[7];
  const float* Wproj = (const float*)d_in[8];
  float* out = (float*)d_out;

  char* p = (char*)d_ws;
  u16* xb  = (u16*)p; p += (size_t)Bc * Tc * Cc * 2;        // 8 MB
  u16* Wqb = (u16*)p; p += (size_t)Cc * Cc * 2;             // 2 MB
  u16* Wkb = (u16*)p; p += (size_t)Cc * Cc * 2;
  u16* Wvb = (u16*)p; p += (size_t)Cc * Cc * 2;
  u16* Wpb = (u16*)p; p += (size_t)Cc * Cc * 2;
  u16* Qb  = (u16*)p; p += (size_t)Bc * Hc * Tc * HDc * 2;  // 8 MB
  u16* Kb  = (u16*)p; p += (size_t)Bc * Hc * SPc * HDc * 2; // 8.5 MB
  u16* Vt  = (u16*)p; p += (size_t)Bc * Hc * HDc * SPc * 2; // 8.5 MB
  u16* Ob  = (u16*)p; p += (size_t)Bc * Tc * Cc * 2;        // 8 MB

  const int nX = Bc * Tc * Cc;            // 4194304
  const int nW = Cc * Cc;                 // 1048576
  const int nK = Bc * Hc * SPc * HDc;     // 4259840

  cvt_bf16<<<nX / 256, 256, 0, stream>>>(x, xb, nX);
  cvtT_bf16<<<nW / 256, 256, 0, stream>>>(Wq, Wqb, Cc, Cc);
  cvtT_bf16<<<nW / 256, 256, 0, stream>>>(Wk, Wkb, Cc, Cc);
  cvtT_bf16<<<nW / 256, 256, 0, stream>>>(Wv, Wvb, Cc, Cc);
  cvtT_bf16<<<nW / 256, 256, 0, stream>>>(Wproj, Wpb, Cc, Cc);

  zero_u16<<<nK / 256, 256, 0, stream>>>(Kb, nK);   // latent-V=0 + pad rows
  zero_u16<<<nK / 256, 256, 0, stream>>>(Vt, nK);

  // (M/16)*(N/64) = 256*16 = 4096 waves, 8 waves/block -> 512 blocks
  gemm_bf16_ep<<<512, 256, 0, stream>>>(xb, Wqb, nullptr, Qb, Bc * Tc, Cc, Cc, 1);
  gemm_bf16_ep<<<512, 256, 0, stream>>>(xb, Wkb, nullptr, Kb, Bc * Tc, Cc, Cc, 2);
  gemm_bf16_ep<<<512, 256, 0, stream>>>(xb, Wvb, nullptr, Vt, Bc * Tc, Cc, Cc, 3);

  latk_kernel<<<(Lc * Cc) / 256, 256, 0, stream>>>(lat_k, Wlk, Kb);

  // B*H*(T/16) = 4096 waves, 4 waves/block -> 1024 blocks of 128 threads
  attn_kernel<<<1024, 128, 0, stream>>>(Qb, Kb, Vt, Ob);

  gemm_bf16_ep<<<512, 256, 0, stream>>>(Ob, Wpb, out, nullptr, Bc * Tc, Cc, Cc, 0);
}
